// GNN_9852654977762
// MI455X (gfx1250) — matrix operand order repaired and checked by reading, not compiled, not run on hardware
//
#include <hip/hip_runtime.h>

#define N_NODES 50000
#define N_EDGES 800000
#define IN_DIM  16
#define HID     64
#define KE      160   // edge-MLP K padded: 130 -> 160 (5 x 32)
#define KN      128   // node-MLP K: 2*HID
#define EDGE_TILES 10
#define NODE_TILES 8

typedef __attribute__((ext_vector_type(16))) __bf16         v16bf;
typedef __attribute__((ext_vector_type(8)))  float          v8f;
typedef __attribute__((ext_vector_type(8)))  unsigned short ushort8;
typedef __attribute__((ext_vector_type(4)))  unsigned short u16x4;
typedef __attribute__((ext_vector_type(4)))  float          f32x4;

union Frag { unsigned short s[16]; v16bf v; };

__device__ __forceinline__ unsigned short f2bf(float f) {
  unsigned int u = __float_as_uint(f);
  u += 0x7FFFu + ((u >> 16) & 1u);     // round-to-nearest-even
  return (unsigned short)(u >> 16);
}

__device__ __forceinline__ u16x4 f4tobf4(f32x4 v) {
  u16x4 r;
  r.x = f2bf(v.x); r.y = f2bf(v.y); r.z = f2bf(v.z); r.w = f2bf(v.w);
  return r;
}

__device__ __forceinline__ float silu_f(float v) {
  return v / (1.0f + __expf(-v));
}

// ---------------------------------------------------------------- embedding
__global__ void __launch_bounds__(256)
gnn_embed(const float* __restrict__ h, const float* __restrict__ w,
          const float* __restrict__ b, float* __restrict__ x) {
  int tid = blockIdx.x * 256 + threadIdx.x;       // 50000*64 threads exactly
  int n = tid >> 6, o = tid & 63;
  if (n >= N_NODES) return;
  float acc = b[o];
#pragma unroll
  for (int k = 0; k < IN_DIM; ++k) acc += h[n * IN_DIM + k] * w[k * HID + o];
  x[n * HID + o] = acc;
}

// ---------------------------------------------------------------- edge MLP + scatter
__global__ void __launch_bounds__(128)
gnn_edge(const float* __restrict__ x, const int* __restrict__ ei,
         const float* __restrict__ ea,
         const float* __restrict__ w1, const float* __restrict__ b1,
         const float* __restrict__ w2, const float* __restrict__ b2,
         float* __restrict__ agg) {
  __shared__ __align__(16) unsigned short sW1t[64 * KE];   // [n][k] transposed
  __shared__ __align__(16) unsigned short sW2t[64 * 64];
  __shared__ float sB1[64], sB2[64];
  __shared__ __align__(16) unsigned short sA[4][16 * KE];  // per-wave A tiles
  __shared__ __align__(16) unsigned short sH[4][16 * 64];  // per-wave hidden
  __shared__ int sRow[4][16], sCol[4][16];

  const int tidb = threadIdx.x;
  for (int idx = tidb; idx < 64 * KE; idx += 128) {
    int n = idx / KE, k = idx - n * KE;
    sW1t[idx] = f2bf(k < 130 ? w1[k * 64 + n] : 0.0f);
  }
  for (int idx = tidb; idx < 64 * 64; idx += 128) {
    int n = idx >> 6, k = idx & 63;
    sW2t[idx] = f2bf(w2[k * 64 + n]);
  }
  if (tidb < 64) { sB1[tidb] = b1[tidb]; sB2[tidb] = b2[tidb]; }

  const int wv = tidb >> 5, ln = tidb & 31;
  const int half = ln >> 4, mrow = ln & 15;

  for (int t = 0; t < EDGE_TILES; ++t) {
    const int ebase = ((blockIdx.x * 4 + wv) * EDGE_TILES + t) * 16;
    __syncthreads();
    if (ln < 16) sRow[wv][ln] = ei[ebase + ln];
    else         sCol[wv][ln - 16] = ei[N_EDGES + ebase + (ln - 16)];
    __syncthreads();
    // ---- gather A tile [x[row] | x[col] | edge_attr | pad] -> bf16, float4-wide
#pragma unroll
    for (int i = 0; i < 8; ++i) {              // x[row] : 16 edges x 16 float4
      int idx = ln + 32 * i;
      int m = idx >> 4, q = idx & 15;
      f32x4 v = *(const f32x4*)&x[sRow[wv][m] * HID + q * 4];
      *(u16x4*)&sA[wv][m * KE + q * 4] = f4tobf4(v);
    }
#pragma unroll
    for (int i = 0; i < 8; ++i) {              // x[col]
      int idx = ln + 32 * i;
      int m = idx >> 4, q = idx & 15;
      f32x4 v = *(const f32x4*)&x[sCol[wv][m] * HID + q * 4];
      *(u16x4*)&sA[wv][m * KE + 64 + q * 4] = f4tobf4(v);
    }
    {                                          // attr + zero pad (k = 128..159)
      ushort8 z = {0, 0, 0, 0, 0, 0, 0, 0};
#pragma unroll
      for (int i = 0; i < 2; ++i) {            // 16 edges x 32 ushorts = 64 x ushort8
        int idx = ln + 32 * i;
        int m = idx >> 2, q = idx & 3;
        *(ushort8*)&sA[wv][m * KE + 128 + q * 8] = z;
      }
      if (ln < 16) {                           // overwrite two attrs (same-wave LDS is issue-ordered)
        float a0 = ea[(ebase + ln) * 2 + 0];
        float a1 = ea[(ebase + ln) * 2 + 1];
        unsigned int packed = (unsigned int)f2bf(a0) | ((unsigned int)f2bf(a1) << 16);
        *(unsigned int*)&sA[wv][ln * KE + 128] = packed;
      }
    }
    __syncthreads();
    // ---- GEMM1: [16 x 160] @ [160 x 64], 5 k-chunks x 4 n-tiles
    v8f acc[4] = {};
    Frag fa, fb;
#pragma unroll
    for (int kc = 0; kc < 5; ++kc) {
      const int ab = mrow * KE + kc * 32 + half * 8;
      *(ushort8*)&fa.s[0] = *(const ushort8*)&sA[wv][ab];
      *(ushort8*)&fa.s[8] = *(const ushort8*)&sA[wv][ab + 16];
#pragma unroll
      for (int nt = 0; nt < 4; ++nt) {
        const int bb = (nt * 16 + mrow) * KE + kc * 32 + half * 8;
        *(ushort8*)&fb.s[0] = *(const ushort8*)&sW1t[bb];
        *(ushort8*)&fb.s[8] = *(const ushort8*)&sW1t[bb + 16];
        acc[nt] = __builtin_amdgcn_wmma_f32_16x16x32_bf16(
            false, fa.v, false, fb.v, (short)0, acc[nt], false, false);
      }
    }
#pragma unroll
    for (int nt = 0; nt < 4; ++nt) {
      const float bias = sB1[nt * 16 + mrow];
#pragma unroll
      for (int r = 0; r < 8; ++r)
        sH[wv][(r + half * 8) * 64 + nt * 16 + mrow] = f2bf(silu_f(acc[nt][r] + bias));
    }
    __syncthreads();
    // ---- GEMM2: [16 x 64] @ [64 x 64]
    v8f acc2[4] = {};
#pragma unroll
    for (int kc = 0; kc < 2; ++kc) {
      const int ab = mrow * 64 + kc * 32 + half * 8;
      *(ushort8*)&fa.s[0] = *(const ushort8*)&sH[wv][ab];
      *(ushort8*)&fa.s[8] = *(const ushort8*)&sH[wv][ab + 16];
#pragma unroll
      for (int nt = 0; nt < 4; ++nt) {
        const int bb = (nt * 16 + mrow) * 64 + kc * 32 + half * 8;
        *(ushort8*)&fb.s[0] = *(const ushort8*)&sW2t[bb];
        *(ushort8*)&fb.s[8] = *(const ushort8*)&sW2t[bb + 16];
        acc2[nt] = __builtin_amdgcn_wmma_f32_16x16x32_bf16(
            false, fa.v, false, fb.v, (short)0, acc2[nt], false, false);
      }
    }
#pragma unroll
    for (int nt = 0; nt < 4; ++nt) {
      const float bias = sB2[nt * 16 + mrow];
#pragma unroll
      for (int r = 0; r < 8; ++r) {
        float v = silu_f(acc2[nt][r] + bias);
        const int m = r + half * 8;
        unsafeAtomicAdd(&agg[sRow[wv][m] * HID + nt * 16 + mrow], v);
      }
    }
  }
}

// ---------------------------------------------------------------- node MLP
__global__ void __launch_bounds__(128)
gnn_node(const float* __restrict__ x, const float* __restrict__ agg,
         const float* __restrict__ w1, const float* __restrict__ b1,
         const float* __restrict__ w2, const float* __restrict__ b2,
         float* __restrict__ xo) {
  __shared__ __align__(16) unsigned short sW1t[64 * KN];
  __shared__ __align__(16) unsigned short sW2t[64 * 64];
  __shared__ float sB1[64], sB2[64];
  __shared__ __align__(16) unsigned short sA[4][16 * KN];
  __shared__ __align__(16) unsigned short sH[4][16 * 64];

  const int tidb = threadIdx.x;
  for (int idx = tidb; idx < 64 * KN; idx += 128) {
    int n = idx >> 7, k = idx & 127;
    sW1t[idx] = f2bf(w1[k * 64 + n]);
  }
  for (int idx = tidb; idx < 64 * 64; idx += 128) {
    int n = idx >> 6, k = idx & 63;
    sW2t[idx] = f2bf(w2[k * 64 + n]);
  }
  if (tidb < 64) { sB1[tidb] = b1[tidb]; sB2[tidb] = b2[tidb]; }

  const int wv = tidb >> 5, ln = tidb & 31;
  const int half = ln >> 4, mrow = ln & 15;

  for (int t = 0; t < NODE_TILES; ++t) {
    int nb = ((blockIdx.x * 4 + wv) * NODE_TILES + t) * 16;
    nb = nb > (N_NODES - 16) ? (N_NODES - 16) : nb;   // uniform clamp (keeps EXEC all-1)
    __syncthreads();
    // ---- gather A tile [x | agg] -> bf16, float4-wide
#pragma unroll
    for (int i = 0; i < 8; ++i) {              // x part
      int idx = ln + 32 * i;
      int m = idx >> 4, q = idx & 15;
      f32x4 v = *(const f32x4*)&x[(nb + m) * HID + q * 4];
      *(u16x4*)&sA[wv][m * KN + q * 4] = f4tobf4(v);
    }
#pragma unroll
    for (int i = 0; i < 8; ++i) {              // agg part
      int idx = ln + 32 * i;
      int m = idx >> 4, q = idx & 15;
      f32x4 v = *(const f32x4*)&agg[(nb + m) * HID + q * 4];
      *(u16x4*)&sA[wv][m * KN + 64 + q * 4] = f4tobf4(v);
    }
    __syncthreads();
    v8f acc[4] = {};
    Frag fa, fb;
#pragma unroll
    for (int kc = 0; kc < 4; ++kc) {
      const int ab = mrow * KN + kc * 32 + half * 8;
      *(ushort8*)&fa.s[0] = *(const ushort8*)&sA[wv][ab];
      *(ushort8*)&fa.s[8] = *(const ushort8*)&sA[wv][ab + 16];
#pragma unroll
      for (int nt = 0; nt < 4; ++nt) {
        const int bb = (nt * 16 + mrow) * KN + kc * 32 + half * 8;
        *(ushort8*)&fb.s[0] = *(const ushort8*)&sW1t[bb];
        *(ushort8*)&fb.s[8] = *(const ushort8*)&sW1t[bb + 16];
        acc[nt] = __builtin_amdgcn_wmma_f32_16x16x32_bf16(
            false, fa.v, false, fb.v, (short)0, acc[nt], false, false);
      }
    }
#pragma unroll
    for (int nt = 0; nt < 4; ++nt) {
      const float bias = sB1[nt * 16 + mrow];
#pragma unroll
      for (int r = 0; r < 8; ++r)
        sH[wv][(r + half * 8) * 64 + nt * 16 + mrow] = f2bf(silu_f(acc[nt][r] + bias));
    }
    __syncthreads();
    v8f acc2[4] = {};
#pragma unroll
    for (int kc = 0; kc < 2; ++kc) {
      const int ab = mrow * 64 + kc * 32 + half * 8;
      *(ushort8*)&fa.s[0] = *(const ushort8*)&sH[wv][ab];
      *(ushort8*)&fa.s[8] = *(const ushort8*)&sH[wv][ab + 16];
#pragma unroll
      for (int nt = 0; nt < 4; ++nt) {
        const int bb = (nt * 16 + mrow) * 64 + kc * 32 + half * 8;
        *(ushort8*)&fb.s[0] = *(const ushort8*)&sW2t[bb];
        *(ushort8*)&fb.s[8] = *(const ushort8*)&sW2t[bb + 16];
        acc2[nt] = __builtin_amdgcn_wmma_f32_16x16x32_bf16(
            false, fa.v, false, fb.v, (short)0, acc2[nt], false, false);
      }
    }
#pragma unroll
    for (int nt = 0; nt < 4; ++nt) {
      const float bias = sB2[nt * 16 + mrow];
#pragma unroll
      for (int r = 0; r < 8; ++r) {
        const int m = r + half * 8;
        xo[(nb + m) * HID + nt * 16 + mrow] = acc2[nt][r] + bias;  // no SiLU on layer-2 out
      }
    }
  }
}

// ---------------------------------------------------------------- decoder
__global__ void __launch_bounds__(64)
gnn_decoder(const float* __restrict__ x, const float* __restrict__ w1,
            const float* __restrict__ b1, const float* __restrict__ w2,
            const float* __restrict__ b2, float* __restrict__ out) {
  __shared__ float sx[64], sh[64];
  const int n = blockIdx.x, t = threadIdx.x;
  sx[t] = x[n * HID + t];
  __syncthreads();
  float a = b1[t];
#pragma unroll 8
  for (int k = 0; k < 64; ++k) a += sx[k] * w1[k * 64 + t];
  sh[t] = silu_f(a);
  __syncthreads();
  if (t < 3) {
    float o = b2[t];
#pragma unroll 8
    for (int k = 0; k < 64; ++k) o += sh[k] * w2[k * 3 + t];
    out[n * 3 + t] = o;
  }
}

// ---------------------------------------------------------------- launcher
extern "C" void kernel_launch(void* const* d_in, const int* in_sizes, int n_in,
                              void* d_out, int out_size, void* d_ws, size_t ws_size,
                              hipStream_t stream) {
  const float* h      = (const float*)d_in[0];
  const int*   ei     = (const int*)  d_in[1];
  const float* ea     = (const float*)d_in[2];
  const float* emb_w  = (const float*)d_in[3];
  const float* emb_b  = (const float*)d_in[4];
  const float* ew1    = (const float*)d_in[5];
  const float* eb1    = (const float*)d_in[6];
  const float* ew2    = (const float*)d_in[7];
  const float* eb2    = (const float*)d_in[8];
  const float* nw1    = (const float*)d_in[9];
  const float* nb1    = (const float*)d_in[10];
  const float* nw2    = (const float*)d_in[11];
  const float* nb2    = (const float*)d_in[12];
  const float* dw1    = (const float*)d_in[13];
  const float* db1    = (const float*)d_in[14];
  const float* dw2    = (const float*)d_in[15];
  const float* db2    = (const float*)d_in[16];

  const size_t NH = (size_t)N_NODES * HID;
  float* xA  = (float*)d_ws;
  float* xB  = xA + NH;
  float* agg = xB + NH;

  gnn_embed<<<(N_NODES * HID) / 256, 256, 0, stream>>>(h, emb_w, emb_b, xA);

  float* xc = xA;
  float* xn = xB;
  const int edge_grid = N_EDGES / (4 * 16 * EDGE_TILES);                 // 1250, exact
  const int node_grid = (N_NODES + 4 * 16 * NODE_TILES - 1) / (4 * 16 * NODE_TILES); // 98

  for (int i = 0; i < 4; ++i) {
    (void)hipMemsetAsync(agg, 0, NH * sizeof(float), stream);
    gnn_edge<<<edge_grid, 128, 0, stream>>>(xc, ei, ea,
        ew1 + (size_t)i * 130 * 64, eb1 + (size_t)i * 64,
        ew2 + (size_t)i * 64 * 64,  eb2 + (size_t)i * 64, agg);
    gnn_node<<<node_grid, 128, 0, stream>>>(xc, agg,
        nw1 + (size_t)i * 128 * 64, nb1 + (size_t)i * 64,
        nw2 + (size_t)i * 64 * 64,  nb2 + (size_t)i * 64, xn);
    float* tmp = xc; xc = xn; xn = tmp;
  }

  gnn_decoder<<<N_NODES, 64, 0, stream>>>(xc, dw1, db1, dw2, db2, (float*)d_out);
}
